// SqueezeAndExcite_72413148610925
// MI455X (gfx1250) — matrix operand order repaired and checked by reading, not compiled, not run on hardware
//
#include <hip/hip_runtime.h>
#include <math.h>

// CDNA5 / gfx1250, wave32. Squeeze-and-Excite over x:(T=32,B=8,C=64,H=64,W=64) f32.
// Memory-bound: ~768 MB total traffic -> ~33us at 23.3 TB/s. MLP done with
// v_wmma_f32_16x16x32_f16 (M=16 pad of B=8, K=32=T, two N-tiles of 16).

typedef __attribute__((ext_vector_type(16))) _Float16 v16h;
typedef __attribute__((ext_vector_type(8)))  float    v8f;

#define T_DIM 32
#define B_DIM 8
#define CHW   262144                 // 64*64*64 elements per (t,b) slab
#define PAIRS (T_DIM * B_DIM)        // 256
#define SPLIT 64                     // partial-sum blocks per slab
#define N4_PER_PAIR  (CHW / 4)       // 65536 float4 per slab
#define N4_PER_CHUNK (N4_PER_PAIR / SPLIT)  // 1024 float4 per block
#define BLOCKS_PER_SLAB 16           // for the scale kernel
#define SCALE_CHUNK4 (N4_PER_PAIR / BLOCKS_PER_SLAB)  // 4096 float4 per block

// ---------------- Phase 1: deterministic split reduction --------------------
__global__ void se_partial_sums(const float* __restrict__ x,
                                float* __restrict__ partials) {
    const int pair  = blockIdx.x >> 6;          // SPLIT == 64
    const int chunk = blockIdx.x & (SPLIT - 1);
    const float4* p = (const float4*)x
                    + (size_t)pair * N4_PER_PAIR
                    + (size_t)chunk * N4_PER_CHUNK;
    float s = 0.f;
    for (int i = threadIdx.x; i < N4_PER_CHUNK; i += 256) {
        float4 v = p[i];                        // global_load_b128
        s += (v.x + v.y) + (v.z + v.w);
    }
    __shared__ float red[256];
    red[threadIdx.x] = s;
    __syncthreads();
    for (int off = 128; off > 0; off >>= 1) {
        if (threadIdx.x < off) red[threadIdx.x] += red[threadIdx.x + off];
        __syncthreads();
    }
    if (threadIdx.x == 0) partials[blockIdx.x] = red[0];  // index = pair*SPLIT+chunk
}

// ---------------- Phase 2: WMMA MLP (one wave32) ----------------------------
__device__ __forceinline__ float sigmoidf_(float v) {
    return 1.0f / (1.0f + __expf(-v));
}

__global__ void se_mlp_wmma(const float* __restrict__ partials,
                            const float* __restrict__ w1,   // (32,32) row-major: w1[j][t]
                            const float* __restrict__ w2,   // (32,32) row-major: w2[i][j]
                            float* __restrict__ gate) {     // out: d[t][b], 32x8
    __shared__ float s_lds[PAIRS];      // s[t][b] = mean over slab
    __shared__ float h_lds[16][T_DIM];  // hidden activations h[b][j] (rows 8..15 == 0)

    const int lane = threadIdx.x;

    // Fold the 64 partials per pair in a fixed order -> deterministic means.
    for (int p = lane * 8; p < lane * 8 + 8; ++p) {
        float acc = 0.f;
        for (int c = 0; c < SPLIT; ++c) acc += partials[p * SPLIT + c];
        s_lds[p] = acc * (1.0f / (float)CHW);
    }
    __syncthreads();

    const int m   = lane & 15;                 // A-layout / C-layout row group
    const int n   = lane & 15;                 // B/C/D-layout column
    const int hiK = (lane >= 16) ? 8 : 0;      // A-layout K offset for upper half-wave
    const int bk  = (lane >= 16) ? 16 : 0;     // unused; B loads use hiK, same k order as A

    // ---- layer 1: H[b][j] = relu( sum_t s[t][b] * w1[j][t] ) ----
    // A[m][k] = s[k][m] (m = b, zero-padded for m>=8), K = t.
    v16h a1;
    #pragma unroll
    for (int e = 0; e < 16; ++e) {
        int k = e + ((e >= 8) ? 8 : 0) + hiK;
        float v = (m < B_DIM) ? s_lds[k * B_DIM + m] : 0.f;
        a1[e] = (_Float16)v;
    }
    // B[k][n] = w1[n + 16*tile][k]
    v16h b1t0, b1t1;
    #pragma unroll
    for (int e = 0; e < 16; ++e) {
        int k = e + ((e >= 8) ? 8 : 0) + hiK;
        b1t0[e] = (_Float16)w1[(n +  0) * T_DIM + k];
        b1t1[e] = (_Float16)w1[(n + 16) * T_DIM + k];
    }
    v8f zero = {};
    v8f h0 = __builtin_amdgcn_wmma_f32_16x16x32_f16(false, a1, false, b1t0,
                                                    (short)0, zero, false, false);
    v8f h1 = __builtin_amdgcn_wmma_f32_16x16x32_f16(false, a1, false, b1t1,
                                                    (short)0, zero, false, false);

    // D layout: vgpr r, lanes>=16 -> M = r+8; N = lane&15. Stash relu(H) to LDS.
    #pragma unroll
    for (int r = 0; r < 8; ++r) {
        int mm = r + ((lane >= 16) ? 8 : 0);
        h_lds[mm][n +  0] = fmaxf(h0[r], 0.f);
        h_lds[mm][n + 16] = fmaxf(h1[r], 0.f);
    }
    __syncthreads();

    // ---- layer 2: G[b][i] = sigmoid( sum_j h[b][j] * w2[i][j] ) ----
    v16h a2;
    #pragma unroll
    for (int e = 0; e < 16; ++e) {
        int k = e + ((e >= 8) ? 8 : 0) + hiK;
        a2[e] = (_Float16)h_lds[m][k];
    }
    v16h b2t0, b2t1;
    #pragma unroll
    for (int e = 0; e < 16; ++e) {
        int k = e + ((e >= 8) ? 8 : 0) + hiK;
        b2t0[e] = (_Float16)w2[(n +  0) * T_DIM + k];
        b2t1[e] = (_Float16)w2[(n + 16) * T_DIM + k];
    }
    v8f g0 = __builtin_amdgcn_wmma_f32_16x16x32_f16(false, a2, false, b2t0,
                                                    (short)0, zero, false, false);
    v8f g1 = __builtin_amdgcn_wmma_f32_16x16x32_f16(false, a2, false, b2t1,
                                                    (short)0, zero, false, false);

    // gate[t][b] = sigmoid(G[b][t]); only lanes 0..15 carry valid b (=r) rows.
    #pragma unroll
    for (int r = 0; r < 8; ++r) {
        int b = r + ((lane >= 16) ? 8 : 0);
        if (b < B_DIM) {
            gate[(n +  0) * B_DIM + b] = sigmoidf_(g0[r]);
            gate[(n + 16) * B_DIM + b] = sigmoidf_(g1[r]);
        }
    }
}

// ---------------- Phase 3: streaming broadcast-scale ------------------------
__global__ void se_scale(const float* __restrict__ x,
                         const float* __restrict__ gate,
                         float* __restrict__ out) {
    const int slab = blockIdx.x >> 4;           // BLOCKS_PER_SLAB == 16
    const float sc = gate[slab];                // block-uniform -> scalar load
    const size_t base = (size_t)blockIdx.x * SCALE_CHUNK4;
    const float4* xin = (const float4*)x + base;
    float4* o = (float4*)out + base;
    for (int i = threadIdx.x; i < SCALE_CHUNK4; i += 256) {
        float4 v = xin[i];                      // global_load_b128
        v.x *= sc; v.y *= sc; v.z *= sc; v.w *= sc;
        o[i] = v;                               // global_store_b128
    }
}

// ---------------------------------------------------------------------------
extern "C" void kernel_launch(void* const* d_in, const int* in_sizes, int n_in,
                              void* d_out, int out_size, void* d_ws, size_t ws_size,
                              hipStream_t stream) {
    (void)in_sizes; (void)n_in; (void)out_size; (void)ws_size;
    const float* x  = (const float*)d_in[0];
    const float* w1 = (const float*)d_in[1];
    const float* w2 = (const float*)d_in[2];
    float* out = (float*)d_out;

    float* partials = (float*)d_ws;                 // PAIRS*SPLIT = 16384 floats
    float* gate     = partials + PAIRS * SPLIT;     // 256 floats

    se_partial_sums<<<PAIRS * SPLIT, 256, 0, stream>>>(x, partials);
    se_mlp_wmma<<<1, 32, 0, stream>>>(partials, w1, w2, gate);
    se_scale<<<PAIRS * BLOCKS_PER_SLAB, 256, 0, stream>>>(x, gate, out);
}
